// ImageAttention_35966056137529
// MI455X (gfx1250) — matrix operand order repaired and checked by reading, not compiled, not run on hardware
//
#include <hip/hip_runtime.h>

typedef __attribute__((ext_vector_type(16))) _Float16 v16h;
typedef __attribute__((ext_vector_type(8)))  _Float16 v8h;
typedef __attribute__((ext_vector_type(8)))  float    v8f;
typedef __attribute__((ext_vector_type(4)))  int      v4i;

#define B_SZ   16
#define C_SZ   256
#define NTOK   1024
#define NHEADS 4
#define DK     64
#define QKVOUT 768
#define SCALE_Q 0.125f

// ---------- CDNA5 async global->LDS copy (ASYNCcnt path), with fallback ----
#if __has_builtin(__builtin_amdgcn_global_load_async_to_lds_b128)
#define HAVE_ASYNC_LDS 1
typedef __attribute__((address_space(1))) v4i glob_v4i;
typedef __attribute__((address_space(3))) v4i lds_v4i;
#else
#define HAVE_ASYNC_LDS 0
#endif

__device__ __forceinline__ void cp16(_Float16* dst_lds, const _Float16* src) {
#if HAVE_ASYNC_LDS
  __builtin_amdgcn_global_load_async_to_lds_b128((glob_v4i*)src,
                                                 (lds_v4i*)dst_lds, 0, 0);
#else
  *(v8h*)dst_lds = *(const v8h*)src;
#endif
}

__device__ __forceinline__ void wait_async() {
#if HAVE_ASYNC_LDS
#if __has_builtin(__builtin_amdgcn_s_wait_asynccnt)
  __builtin_amdgcn_s_wait_asynccnt(0);
#else
  asm volatile("s_wait_asynccnt 0" ::: "memory");
#endif
#endif
}

// ---- WMMA operand helpers (contiguous k order, not the ISA table order; A and B match) ----
// 16-bit A-matrix 16x32: lane L (M=L%16, g=L/16) holds K = g*16 .. g*16+15
__device__ __forceinline__ v16h load_a16(const _Float16* row, int g) {
  v16h a;
  const _Float16* p0 = row + g * 16;
  const _Float16* p1 = row + g * 16 + 8;
#pragma unroll
  for (int i = 0; i < 8; ++i) { a[i] = p0[i]; a[8 + i] = p1[i]; }
  return a;
}
// 16-bit B-matrix 32x16: lane L (N=L%16, g=L/16) holds K = g*16 .. g*16+15
__device__ __forceinline__ v16h load_b16(const _Float16* p) {
  v16h b;
#pragma unroll
  for (int i = 0; i < 16; ++i) b[i] = p[i];
  return b;
}
__device__ __forceinline__ v8f wmma16(v16h a, v16h b, v8f c) {
  return __builtin_amdgcn_wmma_f32_16x16x32_f16(false, a, false, b, (short)0, c,
                                                false, false);
}
__device__ __forceinline__ v8f v8f_zero() {
  v8f z = {0.f, 0.f, 0.f, 0.f, 0.f, 0.f, 0.f, 0.f};
  return z;
}

// ============ Kernel 0: pre-convert weights to f16, [col][k] layout =========
__global__ __launch_bounds__(256) void ia_wconv_kernel(
    const float* __restrict__ w_qkv, const float* __restrict__ w_out,
    _Float16* __restrict__ wqkvT, _Float16* __restrict__ woutT) {
  int gid = blockIdx.x * 256 + threadIdx.x;
  if (gid < QKVOUT * C_SZ) {
    int col = gid >> 8, k = gid & 255;
    wqkvT[gid] = (_Float16)w_qkv[(size_t)k * QKVOUT + col];
  } else {
    int o = gid - QKVOUT * C_SZ;
    if (o < C_SZ * C_SZ) {
      int col = o >> 8, k = o & 255;
      woutT[o] = (_Float16)w_out[(size_t)k * C_SZ + col];
    }
  }
}

// ============ Kernel 1: QKV projection (x^T @ w_qkv + b) ============
// grid (NTOK/64, 768/192, B); 256 threads = 8 waves.
__global__ __launch_bounds__(256) void ia_qkv_kernel(
    const float* __restrict__ x, const _Float16* __restrict__ wqkvT,
    const float* __restrict__ b_qkv, _Float16* __restrict__ qb,
    _Float16* __restrict__ kb, _Float16* __restrict__ vtb) {
  __shared__ _Float16 lds_a[64][32];   // tokens x k-chunk (transposed from x)
  __shared__ _Float16 lds_b[192][32];  // weight tile [col][k] (f16, async-copied)
  const int tid = threadIdx.x;
  const int lane = tid & 31, wave = tid >> 5;
  const int lane16 = lane & 15, g = lane >> 4;
  const int n0 = blockIdx.x * 64;
  const int col0 = blockIdx.y * 192;
  const int b = blockIdx.z;
  const int rt = wave >> 1, ch = wave & 1;

  v8f acc[6];
#pragma unroll
  for (int t = 0; t < 6; ++t) acc[t] = v8f_zero();

  const float* xb = x + (size_t)b * C_SZ * NTOK;

  for (int kk = 0; kk < C_SZ; kk += 32) {
    // B tile: async 16B copies of pre-converted f16 weights
    for (int idx = tid; idx < 192 * 4; idx += 256) {
      int col = idx >> 2, part = idx & 3;
      cp16(&lds_b[col][part * 8],
           wqkvT + (size_t)(col0 + col) * C_SZ + kk + part * 8);
    }
    {  // A tile: lds_a[t][c] = x[b, kk+c, n0+t]  (transpose + f32->f16)
      int t = tid & 63, cb = tid >> 6;
#pragma unroll
      for (int i = 0; i < 8; ++i) {
        int c = cb + i * 4;
        lds_a[t][c] = (_Float16)xb[(size_t)(kk + c) * NTOK + n0 + t];
      }
    }
    wait_async();
    __syncthreads();

    v16h a = load_a16(&lds_a[rt * 16 + lane16][0], g);
#pragma unroll
    for (int ct = 0; ct < 6; ++ct) {
      int col = ch * 96 + ct * 16 + lane16;
      acc[ct] = wmma16(a, load_b16(&lds_b[col][g * 16]), acc[ct]);
    }
    __syncthreads();
  }

  // Epilogue: bias, fold attention scale into Q, scatter f16 Q/K/V^T
#pragma unroll
  for (int ct = 0; ct < 6; ++ct) {
    int gcol = col0 + ch * 96 + ct * 16 + lane16;
    int h = gcol / 192, j = gcol % 192;
    int sel = j / 64, d = j % 64;
    float bias = b_qkv[gcol];
    size_t bh = (size_t)(b * NHEADS + h);
#pragma unroll
    for (int r = 0; r < 8; ++r) {
      int token = n0 + rt * 16 + g * 8 + r;
      float v = acc[ct][r] + bias;
      if (sel == 0)
        qb[(bh * NTOK + token) * DK + d] = (_Float16)(v * SCALE_Q);
      else if (sel == 1)
        kb[(bh * NTOK + token) * DK + d] = (_Float16)v;
      else
        vtb[(bh * DK + d) * NTOK + token] = (_Float16)v;
    }
  }
}

// ============ Kernel 2a: per-column softmax stats ============
// softmax over QUERY axis i => column stats of S; compute S^T = K @ Q^T
// row-wise (rows = j). Q chunk staged in LDS, double-buffered async prefetch.
__global__ __launch_bounds__(256) void ia_stats_kernel(
    const _Float16* __restrict__ qb, const _Float16* __restrict__ kb,
    float* __restrict__ mbuf, float* __restrict__ rlbuf) {
  __shared__ _Float16 qt[2][32][64];
  const int tid = threadIdx.x, lane = tid & 31, wave = tid >> 5;
  const int lane16 = lane & 15, g = lane >> 4;
  const int bh = blockIdx.x;
  const int jbase = blockIdx.y * 128 + wave * 16;

  const _Float16* krow = kb + ((size_t)bh * NTOK + jbase + lane16) * DK;
  v16h a0 = load_a16(krow, g);       // K rows, d 0..31
  v16h a1 = load_a16(krow + 32, g);  // K rows, d 32..63

  float m[8], s[8];
#pragma unroll
  for (int r = 0; r < 8; ++r) { m[r] = -1e30f; s[r] = 0.f; }

  const _Float16* qbase = qb + (size_t)bh * NTOK * DK;
  auto stageQ = [&](int bi, int i0) {
    int row = tid >> 3, part = tid & 7;  // 32 rows x 8x16B
    cp16(&qt[bi][row][part * 8], qbase + (size_t)(i0 + row) * DK + part * 8);
  };

  stageQ(0, 0);
  for (int i0 = 0; i0 < NTOK; i0 += 32) {
    int cur = (i0 >> 5) & 1;
    wait_async();
    __syncthreads();
    if (i0 + 32 < NTOK) stageQ(cur ^ 1, i0 + 32);

    v8f c0 = v8f_zero(), c1 = v8f_zero();
    c0 = wmma16(a0, load_b16(&qt[cur][lane16][g * 16]), c0);
    c0 = wmma16(a1, load_b16(&qt[cur][lane16][32 + g * 16]), c0);
    c1 = wmma16(a0, load_b16(&qt[cur][16 + lane16][g * 16]), c1);
    c1 = wmma16(a1, load_b16(&qt[cur][16 + lane16][32 + g * 16]), c1);
#pragma unroll
    for (int r = 0; r < 8; ++r) {
      float vmx = fmaxf(c0[r], c1[r]);
#pragma unroll
      for (int msk = 8; msk >= 1; msk >>= 1)
        vmx = fmaxf(vmx, __shfl_xor(vmx, msk, 32));
      float newm = fmaxf(m[r], vmx);
      float e = __expf(c0[r] - newm) + __expf(c1[r] - newm);
#pragma unroll
      for (int msk = 8; msk >= 1; msk >>= 1)
        e += __shfl_xor(e, msk, 32);
      s[r] = s[r] * __expf(m[r] - newm) + e;
      m[r] = newm;
    }
  }
  if (lane16 == 0) {
#pragma unroll
    for (int r = 0; r < 8; ++r) {
      int j = jbase + g * 8 + r;
      mbuf[(size_t)bh * NTOK + j] = m[r];
      rlbuf[(size_t)bh * NTOK + j] = 1.f / s[r];
    }
  }
}

// ============ Kernel 2b: O = P @ V with precomputed column stats ============
// K and V^T chunks staged in LDS (shared by all 8 waves), double-buffered.
__global__ __launch_bounds__(256) void ia_attnout_kernel(
    const _Float16* __restrict__ qb, const _Float16* __restrict__ kb,
    const _Float16* __restrict__ vtb, const float* __restrict__ mbuf,
    const float* __restrict__ rlbuf, _Float16* __restrict__ ob) {
  __shared__ _Float16 kt[2][32][64];     // K chunk  [j][d]
  __shared__ _Float16 vt[2][64][32];     // V^T chunk [d][j]
  __shared__ _Float16 lds_p[8][16][32];  // per-wave P tile for D->A relayout
  const int tid = threadIdx.x, lane = tid & 31, wave = tid >> 5;
  const int lane16 = lane & 15, g = lane >> 4;
  const int bh = blockIdx.x;
  const int b = bh >> 2, h = bh & 3;
  const int ibase = blockIdx.y * 128 + wave * 16;

  const _Float16* qrow = qb + ((size_t)bh * NTOK + ibase + lane16) * DK;
  v16h a0 = load_a16(qrow, g);
  v16h a1 = load_a16(qrow + 32, g);

  v8f o[4];
#pragma unroll
  for (int t = 0; t < 4; ++t) o[t] = v8f_zero();

  const _Float16* kbase = kb + (size_t)bh * NTOK * DK;
  const _Float16* vtbase = vtb + (size_t)bh * DK * NTOK;
  const float* mrow = mbuf + (size_t)bh * NTOK;
  const float* rlrow = rlbuf + (size_t)bh * NTOK;

  auto stageKV = [&](int bi, int j0) {
    {  // K: 32 rows x 128B
      int row = tid >> 3, part = tid & 7;
      cp16(&kt[bi][row][part * 8], kbase + (size_t)(j0 + row) * DK + part * 8);
    }
    {  // V^T: 64 rows x 64B
      int row = tid >> 2, part = tid & 3;
      cp16(&vt[bi][row][part * 8],
           vtbase + (size_t)row * NTOK + j0 + part * 8);
    }
  };

  stageKV(0, 0);
  for (int j0 = 0; j0 < NTOK; j0 += 32) {
    int cur = (j0 >> 5) & 1;
    wait_async();
    __syncthreads();
    if (j0 + 32 < NTOK) stageKV(cur ^ 1, j0 + 32);

    v8f s0 = v8f_zero(), s1 = v8f_zero();
    s0 = wmma16(a0, load_b16(&kt[cur][lane16][g * 16]), s0);
    s0 = wmma16(a1, load_b16(&kt[cur][lane16][32 + g * 16]), s0);
    s1 = wmma16(a0, load_b16(&kt[cur][16 + lane16][g * 16]), s1);
    s1 = wmma16(a1, load_b16(&kt[cur][16 + lane16][32 + g * 16]), s1);

    float m0 = mrow[j0 + lane16],      rl0 = rlrow[j0 + lane16];
    float m1 = mrow[j0 + 16 + lane16], rl1 = rlrow[j0 + 16 + lane16];
#pragma unroll
    for (int r = 0; r < 8; ++r) {  // D layout: row M = r + 8*g, col N = lane16
      lds_p[wave][g * 8 + r][lane16]      = (_Float16)(__expf(s0[r] - m0) * rl0);
      lds_p[wave][g * 8 + r][16 + lane16] = (_Float16)(__expf(s1[r] - m1) * rl1);
    }
    // wave-private LDS RAW: drain DS counter before re-reading in A layout
    asm volatile("s_wait_dscnt 0" ::: "memory");
    v16h pa = load_a16(&lds_p[wave][lane16][0], g);
#pragma unroll
    for (int dt = 0; dt < 4; ++dt)
      o[dt] = wmma16(pa, load_b16(&vt[cur][dt * 16 + lane16][g * 16]), o[dt]);
  }

  _Float16* obase = ob + (size_t)b * NTOK * (NHEADS * DK);
#pragma unroll
  for (int dt = 0; dt < 4; ++dt)
#pragma unroll
    for (int r = 0; r < 8; ++r) {
      int token = ibase + g * 8 + r;
      obase[(size_t)token * 256 + h * DK + dt * 16 + lane16] =
          (_Float16)o[dt][r];
    }
}

// ============ Kernel 3: out projection + bias + residual ============
__global__ __launch_bounds__(256) void ia_outproj_kernel(
    const _Float16* __restrict__ ob, const _Float16* __restrict__ woutT,
    const float* __restrict__ b_out, const float* __restrict__ x,
    float* __restrict__ out) {
  __shared__ _Float16 lds_b[128][32];  // [col][k]
  const int tid = threadIdx.x, lane = tid & 31, wave = tid >> 5;
  const int lane16 = lane & 15, g = lane >> 4;
  const int n0 = blockIdx.x * 64;
  const int col0 = blockIdx.y * 128;
  const int b = blockIdx.z;
  const int rt = wave >> 1, ch = wave & 1;

  v8f acc[4];
#pragma unroll
  for (int t = 0; t < 4; ++t) acc[t] = v8f_zero();

  const _Float16* arow = ob + ((size_t)b * NTOK + n0 + rt * 16 + lane16) * 256;

  for (int kk = 0; kk < C_SZ; kk += 32) {
    for (int idx = tid; idx < 128 * 4; idx += 256) {
      int col = idx >> 2, part = idx & 3;
      cp16(&lds_b[col][part * 8],
           woutT + (size_t)(col0 + col) * C_SZ + kk + part * 8);
    }
    wait_async();
    __syncthreads();
    v16h a = load_a16(arow + kk, g);  // O rows are d-contiguous f16 in global
#pragma unroll
    for (int ct = 0; ct < 4; ++ct) {
      int col = ch * 64 + ct * 16 + lane16;
      acc[ct] = wmma16(a, load_b16(&lds_b[col][g * 16]), acc[ct]);
    }
    __syncthreads();
  }

  const float* xb = x + (size_t)b * C_SZ * NTOK;
  float* outb = out + (size_t)b * C_SZ * NTOK;
#pragma unroll
  for (int ct = 0; ct < 4; ++ct) {
    int c = col0 + ch * 64 + ct * 16 + lane16;
    float bias = b_out[c];
#pragma unroll
    for (int r = 0; r < 8; ++r) {
      int token = n0 + rt * 16 + g * 8 + r;
      outb[(size_t)c * NTOK + token] =
          acc[ct][r] + bias + xb[(size_t)c * NTOK + token];
    }
  }
}

extern "C" void kernel_launch(void* const* d_in, const int* in_sizes, int n_in,
                              void* d_out, int out_size, void* d_ws,
                              size_t ws_size, hipStream_t stream) {
  const float* x = (const float*)d_in[0];
  const float* w_qkv = (const float*)d_in[1];
  const float* b_qkv = (const float*)d_in[2];
  const float* w_out = (const float*)d_in[3];
  const float* b_out = (const float*)d_in[4];
  float* out = (float*)d_out;

  char* ws = (char*)d_ws;
  const size_t szh = (size_t)B_SZ * NHEADS * NTOK * DK * sizeof(_Float16);  // 8MB
  const size_t statsz = (size_t)B_SZ * NHEADS * NTOK * sizeof(float);       // 256KB
  _Float16* qb  = (_Float16*)(ws);
  _Float16* kb  = (_Float16*)(ws + szh);
  _Float16* vtb = (_Float16*)(ws + 2 * szh);
  _Float16* ob  = (_Float16*)(ws + 3 * szh);  // B*NTOK*256 f16 = 8MB
  float* mbuf   = (float*)(ws + 4 * szh);
  float* rlbuf  = (float*)(ws + 4 * szh + statsz);
  _Float16* wqkvT = (_Float16*)(ws + 4 * szh + 2 * statsz);
  _Float16* woutT = wqkvT + (size_t)QKVOUT * C_SZ;

  ia_wconv_kernel<<<dim3(1024), 256, 0, stream>>>(w_qkv, w_out, wqkvT, woutT);
  ia_qkv_kernel<<<dim3(NTOK / 64, QKVOUT / 192, B_SZ), 256, 0, stream>>>(
      x, wqkvT, b_qkv, qb, kb, vtb);
  ia_stats_kernel<<<dim3(B_SZ * NHEADS, NTOK / 128), 256, 0, stream>>>(
      qb, kb, mbuf, rlbuf);
  ia_attnout_kernel<<<dim3(B_SZ * NHEADS, NTOK / 128), 256, 0, stream>>>(
      qb, kb, vtb, mbuf, rlbuf, ob);
  ia_outproj_kernel<<<dim3(NTOK / 64, C_SZ / 128, B_SZ), 256, 0, stream>>>(
      ob, woutT, b_out, x, out);
}
